// MultiHeadAttention_37615323578654
// MI455X (gfx1250) — matrix operand order repaired and checked by reading, not compiled, not run on hardware
//
#include <hip/hip_runtime.h>
#include <hip/hip_bf16.h>

#define D_MODEL 512
#define N_HEADS 8
#define D_K     64
#define SEQ     2048
#define BATCH   2
#define NTOK    (BATCH * SEQ)  // 4096

typedef __attribute__((ext_vector_type(16))) __bf16 v16bf;
typedef __attribute__((ext_vector_type(8)))  __bf16 v8bf;
typedef __attribute__((ext_vector_type(4)))  __bf16 v4bf;
typedef __attribute__((ext_vector_type(8)))  float  v8f;
typedef __attribute__((ext_vector_type(4)))  float  v4f;

__device__ __forceinline__ v8f wmma_bf16(v16bf a, v16bf b, v8f c) {
  // emits v_wmma_f32_16x16x32_bf16
  return __builtin_amdgcn_wmma_f32_16x16x32_bf16(false, a, false, b, (short)0, c,
                                                 false, false);
}

// ---- WMMA operand loaders (contiguous k order shared by A and B, wave32) ----
// A (16x32, MxK), row-major source. Lane L: row = L&15,
// k = 16*hi + e (16 contiguous elements), hi = L>>4.
__device__ __forceinline__ v16bf load_a_bf16(const __bf16* p, int ld) {
  const int lane = threadIdx.x & 31;
  const __bf16* r = p + (size_t)(lane & 15) * ld + ((lane >> 4) << 4);
  v8bf lo = *(const v8bf*)r;
  v8bf hi = *(const v8bf*)(r + 8);
  return __builtin_shufflevector(lo, hi, 0, 1, 2, 3, 4, 5, 6, 7, 8, 9, 10, 11,
                                 12, 13, 14, 15);
}

__device__ __forceinline__ v16bf load_a_f32(const float* p, int ld) {
  const int lane = threadIdx.x & 31;
  const float* r = p + (size_t)(lane & 15) * ld + ((lane >> 4) << 4);
  v8f lo = *(const v8f*)r;
  v8f hi = *(const v8f*)(r + 8);
  v16bf a;
#pragma unroll
  for (int i = 0; i < 8; ++i) { a[i] = (__bf16)lo[i]; a[8 + i] = (__bf16)hi[i]; }
  return a;
}

// B (32x16, KxN) with B(k,n) = src[n*ld + k] (n-major source, e.g. W[n][k]).
// Lane: n = lane&15, k = 16*(lane>>4) + e -> 16 contiguous elements.
__device__ __forceinline__ v16bf load_b_bf16(const __bf16* p, int ld) {
  const int lane = threadIdx.x & 31;
  const __bf16* r = p + (size_t)(lane & 15) * ld + ((lane >> 4) << 4);
  return *(const v16bf*)r;
}

__device__ __forceinline__ v16bf load_b_f32(const float* p, int ld) {
  const int lane = threadIdx.x & 31;
  const float* r = p + (size_t)(lane & 15) * ld + ((lane >> 4) << 4);
  v8f lo = *(const v8f*)r;
  v8f hi = *(const v8f*)(r + 8);
  v16bf b;
#pragma unroll
  for (int i = 0; i < 8; ++i) { b[i] = (__bf16)lo[i]; b[8 + i] = (__bf16)hi[i]; }
  return b;
}

// ---------------- Kernel 1: fused Q/K/V projections ----------------
// grid = (NTOK/16, N_HEADS, 3); one wave computes a 16x64 output tile of
// X @ W^T + b and scatters into per-head bf16 layouts.
__global__ __launch_bounds__(32) void qkv_proj_kernel(
    const float* __restrict__ Q, const float* __restrict__ K,
    const float* __restrict__ V, const float* __restrict__ Wq,
    const float* __restrict__ bq, const float* __restrict__ Wk,
    const float* __restrict__ bk, const float* __restrict__ Wv,
    const float* __restrict__ bv, __bf16* __restrict__ q_ws,
    __bf16* __restrict__ k_ws, __bf16* __restrict__ vT_ws) {
  const int sel = blockIdx.z;
  const float* X    = (sel == 0) ? Q : (sel == 1) ? K : V;
  const float* W    = (sel == 0) ? Wq : (sel == 1) ? Wk : Wv;
  const float* bias = (sel == 0) ? bq : (sel == 1) ? bk : bv;

  const int m0 = blockIdx.x * 16;   // token tile
  const int h  = blockIdx.y;        // head == 64-wide feature block
  const int n0b = h * D_K;
  const int lane = threadIdx.x & 31;
  const int hi = lane >> 4, nn = lane & 15;

  v8f c[4] = {};
  for (int k0 = 0; k0 < D_MODEL; k0 += 32) {
    v16bf a = load_a_f32(X + (size_t)m0 * D_MODEL + k0, D_MODEL);
#pragma unroll
    for (int t = 0; t < 4; ++t) {
      v16bf bm = load_b_f32(W + (size_t)(n0b + 16 * t) * D_MODEL + k0, D_MODEL);
      c[t] = wmma_bf16(a, bm, c[t]);
    }
  }
#pragma unroll
  for (int t = 0; t < 4; ++t) {
    const int d0 = 16 * t + nn;                    // feature within head
    const float bval = bias[n0b + d0];
#pragma unroll
    for (int e = 0; e < 8; ++e) {
      const int tok = m0 + e + 8 * hi;
      const int b = tok >> 11, s = tok & (SEQ - 1);
      const float val = c[t][e] + bval;
      if (sel < 2) {
        __bf16* dst = (sel == 0) ? q_ws : k_ws;   // [b,h,s,dk]
        dst[(((size_t)(b * N_HEADS + h)) * SEQ + s) * D_K + d0] = (__bf16)val;
      } else {                                     // v transposed: [b,h,dk,s]
        vT_ws[(((size_t)(b * N_HEADS + h)) * D_K + d0) * SEQ + s] = (__bf16)val;
      }
    }
  }
}

// ---------------- Kernel 2: attention (scores + softmax + attn@V) ----------
// grid = BATCH*N_HEADS*(SEQ/16); one wave owns 16 query rows.
// Dynamic LDS: 16x2048 f32 scores (128KB, later overlaid in place with bf16
// probabilities) + 16 row maxima. attn tensor is written to HBM exactly once.
__global__ __launch_bounds__(32) void attn_kernel(
    const __bf16* __restrict__ q_ws, const __bf16* __restrict__ k_ws,
    const __bf16* __restrict__ vT_ws, __bf16* __restrict__ outh_ws,
    float* __restrict__ attn_out) {
  extern __shared__ char smem[];
  float* sc = (float*)smem;                                // [16][SEQ] f32
  float* rowmax = (float*)(smem + (size_t)16 * SEQ * 4);   // [16]
  // bf16 probability overlay: row r starts at the f32 row slot, so the bf16
  // row stride is 2*SEQ elements (row fits in the first half of the slot).
  __bf16* probs = (__bf16*)smem;

  const int blk = blockIdx.x;
  const int qblk = blk & (SEQ / 16 - 1);  // 0..127
  const int bh   = blk >> 7;              // b*8+h, 0..15
  const int s0   = qblk * 16;
  const int lane = threadIdx.x & 31;
  const int hi = lane >> 4, nn = lane & 15;

  const __bf16* qb  = q_ws  + ((size_t)bh * SEQ + s0) * D_K;
  const __bf16* kb  = k_ws  + (size_t)bh * SEQ * D_K;
  const __bf16* vtb = vT_ws + (size_t)bh * D_K * SEQ;
  float* attn_bh = attn_out + (size_t)bh * SEQ * SEQ + (size_t)s0 * SEQ;

  // Q tile A-operands, reused across all 128 key tiles.
  v16bf aq0 = load_a_bf16(qb, D_K);
  v16bf aq1 = load_a_bf16(qb + 32, D_K);

  float rmax[8];
#pragma unroll
  for (int e = 0; e < 8; ++e) rmax[e] = -3.0e38f;

  // Pass 1: scaled scores -> LDS (f32), track row maxima. No global stores.
  for (int n0 = 0; n0 < SEQ; n0 += 16) {
    v16bf b0 = load_b_bf16(kb + (size_t)n0 * D_K, D_K);
    v16bf b1 = load_b_bf16(kb + (size_t)n0 * D_K + 32, D_K);
    v8f c = {};
    c = wmma_bf16(aq0, b0, c);
    c = wmma_bf16(aq1, b1, c);
#pragma unroll
    for (int e = 0; e < 8; ++e) {
      const float v = c[e] * 0.125f;  // 1/sqrt(64)
      rmax[e] = fmaxf(rmax[e], v);
      sc[(size_t)(e + 8 * hi) * SEQ + n0 + nn] = v;
    }
  }
  // reduce maxima across the 16 lanes of each half-wave
#pragma unroll
  for (int e = 0; e < 8; ++e) {
    float v = rmax[e];
    v = fmaxf(v, __shfl_xor(v, 1, 32));
    v = fmaxf(v, __shfl_xor(v, 2, 32));
    v = fmaxf(v, __shfl_xor(v, 4, 32));
    v = fmaxf(v, __shfl_xor(v, 8, 32));
    rmax[e] = v;
  }
  if (nn == 0) {
#pragma unroll
    for (int e = 0; e < 8; ++e) rowmax[e + 8 * hi] = rmax[e];
  }
  __syncthreads();

  // Pass 2: per-row softmax in fp32 from LDS. Normalized fp32 attn is written
  // to global exactly once (b128 stores); bf16 probs overwrite the LDS row.
  for (int r = 0; r < 16; ++r) {
    const float m = rowmax[r];
    const v4f* srow = (const v4f*)(sc + (size_t)r * SEQ);
    v4f ev[SEQ / 128];
    float psum = 0.f;
#pragma unroll
    for (int i = 0; i < SEQ / 128; ++i) {  // 2048/4 float4s / 32 lanes
      v4f x = srow[lane + 32 * i];
      v4f e4;
#pragma unroll
      for (int j = 0; j < 4; ++j) { e4[j] = __expf(x[j] - m); psum += e4[j]; }
      ev[i] = e4;
    }
    psum += __shfl_xor(psum, 1, 32);
    psum += __shfl_xor(psum, 2, 32);
    psum += __shfl_xor(psum, 4, 32);
    psum += __shfl_xor(psum, 8, 32);
    psum += __shfl_xor(psum, 16, 32);
    const float inv = 1.0f / psum;

    v4f* growp = (v4f*)(attn_bh + (size_t)r * SEQ);
    v4bf* prow = (v4bf*)(probs + (size_t)r * 2 * SEQ);
#pragma unroll
    for (int i = 0; i < SEQ / 128; ++i) {
      v4f p = ev[i];
      v4bf pb;
#pragma unroll
      for (int j = 0; j < 4; ++j) { p[j] *= inv; pb[j] = (__bf16)p[j]; }
      growp[lane + 32 * i] = p;   // global_store_b128 (the only attn write)
      prow[lane + 32 * i] = pb;   // ds_store_b64, overlays f32 row in place
    }
  }
  __syncthreads();

  // Pass 3: out_heads(16 x 64) = probs(16 x 2048 bf16, LDS) @ V(2048 x 64).
  const int bb = bh >> 3, hh = bh & 7;
#pragma unroll
  for (int t = 0; t < 4; ++t) {
    const int d0 = 16 * t;
    v8f c = {};
    for (int kk = 0; kk < SEQ; kk += 32) {
      v16bf a  = load_a_bf16(probs + kk, 2 * SEQ);                 // ds loads
      v16bf bm = load_b_bf16(vtb + (size_t)d0 * SEQ + kk, SEQ);
      c = wmma_bf16(a, bm, c);
    }
#pragma unroll
    for (int e = 0; e < 8; ++e) {
      const size_t tok = (size_t)bb * SEQ + s0 + e + 8 * hi;
      outh_ws[tok * D_MODEL + hh * D_K + d0 + nn] = (__bf16)c[e];
    }
  }
}

// ---------------- Kernel 3: output projection ----------------
// grid = (NTOK/16, D_MODEL/64); out = out_heads @ Wo^T + bo (fp32 result).
__global__ __launch_bounds__(32) void out_proj_kernel(
    const __bf16* __restrict__ outh, const float* __restrict__ Wo,
    const float* __restrict__ bo, float* __restrict__ out) {
  const int m0 = blockIdx.x * 16;
  const int n0b = blockIdx.y * 64;
  const int lane = threadIdx.x & 31;
  const int hi = lane >> 4, nn = lane & 15;

  v8f c[4] = {};
  for (int k0 = 0; k0 < D_MODEL; k0 += 32) {
    v16bf a = load_a_bf16(outh + (size_t)m0 * D_MODEL + k0, D_MODEL);
#pragma unroll
    for (int t = 0; t < 4; ++t) {
      v16bf bm = load_b_f32(Wo + (size_t)(n0b + 16 * t) * D_MODEL + k0, D_MODEL);
      c[t] = wmma_bf16(a, bm, c[t]);
    }
  }
#pragma unroll
  for (int t = 0; t < 4; ++t) {
    const float bval = bo[n0b + 16 * t + nn];
#pragma unroll
    for (int e = 0; e < 8; ++e) {
      out[(size_t)(m0 + e + 8 * hi) * D_MODEL + n0b + 16 * t + nn] =
          c[t][e] + bval;
    }
  }
}

extern "C" void kernel_launch(void* const* d_in, const int* in_sizes, int n_in,
                              void* d_out, int out_size, void* d_ws,
                              size_t ws_size, hipStream_t stream) {
  const float* Q  = (const float*)d_in[0];
  const float* K  = (const float*)d_in[1];
  const float* V  = (const float*)d_in[2];
  const float* Wq = (const float*)d_in[3];
  const float* bq = (const float*)d_in[4];
  const float* Wk = (const float*)d_in[5];
  const float* bk = (const float*)d_in[6];
  const float* Wv = (const float*)d_in[7];
  const float* bv = (const float*)d_in[8];
  const float* Wo = (const float*)d_in[9];
  const float* bo = (const float*)d_in[10];

  float* out  = (float*)d_out;                       // [4096,512]
  float* attn = out + (size_t)NTOK * D_MODEL;        // [2,8,2048,2048]

  // Workspace layout (bf16): q 4MB | k 4MB | vT 4MB | out_heads 4MB = 16MB.
  char* ws = (char*)d_ws;
  __bf16* q_ws  = (__bf16*)(ws);
  __bf16* k_ws  = (__bf16*)(ws + ((size_t)4 << 20));
  __bf16* vT_ws = (__bf16*)(ws + ((size_t)8 << 20));
  __bf16* outh  = (__bf16*)(ws + ((size_t)12 << 20));

  qkv_proj_kernel<<<dim3(NTOK / 16, N_HEADS, 3), 32, 0, stream>>>(
      Q, K, V, Wq, bq, Wk, bk, Wv, bv, q_ws, k_ws, vT_ws);

  const size_t attn_lds =
      (size_t)16 * SEQ * sizeof(float) + 16 * sizeof(float);  // 128KB + 64B
  attn_kernel<<<dim3(BATCH * N_HEADS * (SEQ / 16)), 32, attn_lds, stream>>>(
      q_ws, k_ws, vT_ws, outh, attn);

  out_proj_kernel<<<dim3(NTOK / 16, D_MODEL / 64), 32, 0, stream>>>(
      outh, Wo, bo, out);
}
